// PiCANet_1932735283852
// MI455X (gfx1250) — matrix operand order repaired and checked by reading, not compiled, not run on hardware
//
#include <hip/hip_runtime.h>
#include <hip/hip_bf16.h>

// ---------------------------------------------------------------------------
// PiCANet-L on MI455X (gfx1250).
// Conv1 (dilated 7x7, 256->128, ~161 GFLOP) dominates: implicit GEMM on the
// bf16 WMMA pipe with a zero-padded halo layout so the MAC loop has NO bounds
// checks / divides, double-buffered LDS A-tiles shared by 8 waves, and 4
// accumulators per wave for XDL pipelining. A-tile staging uses the gfx1250
// async global->LDS path (ASYNCcnt) when the toolchain provides it.
// ---------------------------------------------------------------------------

typedef __attribute__((ext_vector_type(16))) __bf16 v16bf;
typedef __attribute__((ext_vector_type(8)))  __bf16 v8bf;
typedef __attribute__((ext_vector_type(8)))  float  v8f;
typedef int v4i_vs __attribute__((vector_size(16)));   // matches builtin proto

constexpr int BATCH = 16;
constexpr int CIN   = 256;
constexpr int H     = 56;
constexpr int W     = 56;
constexpr int HW    = H * W;            // 3136
constexpr int NPIX  = BATCH * HW;       // 50176
constexpr int CMID  = 128;
constexpr int NTAP  = 49;
constexpr int NATT  = 64;               // 49 padded to 64
constexpr float BN_EPS = 1e-5f;

// Halo-padded x: h in [-6,61] -> hh=h+6 in [0,68); w in [-8,71] -> ww=w+8 in [0,80)
constexpr int HP = 68;
constexpr int WP = 80;
constexpr int MBLK = 64;                // pixels (padded w window) per block

// ---------------- workspace layout (bytes, all 256B aligned) ----------------
constexpr size_t OFF_XH  = 0;                                    // [B][HP][WP][C] bf16
constexpr size_t SZ_XH   = (size_t)BATCH * HP * WP * CIN * 2;    // 44,564,480
constexpr size_t OFF_W1B = OFF_XH + SZ_XH;                       // [tap][o][c] bf16
constexpr size_t SZ_W1B  = (size_t)NTAP * CMID * CIN * 2;        //  3,211,264
constexpr size_t OFF_W2B = OFF_W1B + SZ_W1B;                     // [64][128] bf16
constexpr size_t SZ_W2B  = (size_t)NATT * CMID * 2;
constexpr size_t OFF_K1B = OFF_W2B + SZ_W2B;                     // [pix][128] bf16
constexpr size_t SZ_K1B  = (size_t)NPIX * CMID * 2;              // 12,845,056
constexpr size_t OFF_LOG = OFF_K1B + SZ_K1B;                     // [pix][64] f32 (softmax in-place)
constexpr size_t SZ_LOG  = (size_t)NPIX * NATT * 4;              // 12,845,056
// total ~73.5 MB

// ---------------- gfx1250 async global->LDS path (guarded) -----------------
#if defined(__has_builtin)
# if __has_builtin(__builtin_amdgcn_global_load_async_to_lds_b128)
#  define ASYNC_LDS 1
# endif
#endif

template <int N>
__device__ __forceinline__ void wait_async_lds()
{
#if defined(ASYNC_LDS)
# if __has_builtin(__builtin_amdgcn_s_wait_asynccnt)
    __builtin_amdgcn_s_wait_asynccnt((short)N);
# else
    asm volatile("s_wait_asynccnt %0" :: "i"(N) : "memory");
# endif
#endif
}

// Copy one contiguous 32KB tap tile (64 px x 256 ch bf16) global -> LDS.
// 256 threads x 8 passes x 16B.
__device__ __forceinline__ void stage_tile(const __bf16* __restrict__ gsrc,
                                           __bf16* lds_dst, int tid)
{
#pragma unroll
    for (int pass = 0; pass < 8; ++pass) {
        const int off = (pass * 256 + tid) * 8;   // elements (8 bf16 = 16B)
#if defined(ASYNC_LDS)
        __builtin_amdgcn_global_load_async_to_lds_b128(
            (__attribute__((address_space(1))) v4i_vs*)(gsrc + off),
            (__attribute__((address_space(3))) v4i_vs*)(lds_dst + off),
            0, 0);
#else
        *(v8bf*)(lds_dst + off) = *(const v8bf*)(gsrc + off);
#endif
    }
}

// ---------------------------------------------------------------------------
// Kernel 1: x NCHW f32 -> halo-padded NHWC bf16 (zeros in halo)
// ---------------------------------------------------------------------------
__global__ __launch_bounds__(256) void pack_x_halo_bf16(
    const float* __restrict__ x, __bf16* __restrict__ xh)
{
    int tid = blockIdx.x * 256 + threadIdx.x;     // (b,hh,ww,c)
    int c   = tid & 255;
    int pix = tid >> 8;
    int ww  = pix % WP;
    int hh  = (pix / WP) % HP;
    int b   = pix / (WP * HP);
    int h   = hh - 6;
    int w   = ww - 8;
    float v = 0.0f;
    if (h >= 0 && h < H && w >= 0 && w < W)
        v = x[((size_t)(b * CIN + c)) * HW + h * W + w];
    xh[tid] = (__bf16)v;
}

// ---------------------------------------------------------------------------
// Kernel 2: w1 (128,256,7,7) f32 -> [tap][o][c] bf16 (c contiguous)
// ---------------------------------------------------------------------------
__global__ __launch_bounds__(256) void pack_w1_bf16(
    const float* __restrict__ w1, __bf16* __restrict__ w1b)
{
    int tid = blockIdx.x * 256 + threadIdx.x;     // t*(128*256) + o*256 + c
    int t   = tid / (CMID * CIN);
    int rem = tid % (CMID * CIN);
    int o   = rem >> 8;
    int c   = rem & 255;
    w1b[tid] = (__bf16)w1[((size_t)(o * CIN + c)) * NTAP + t];
}

// ---------------------------------------------------------------------------
// Kernel 3: w2 (49,128) f32 -> [64][128] bf16 zero-padded
// ---------------------------------------------------------------------------
__global__ __launch_bounds__(256) void pack_w2_bf16(
    const float* __restrict__ w2, __bf16* __restrict__ w2b)
{
    int tid = blockIdx.x * 256 + threadIdx.x;     // o*128 + c
    int o   = tid >> 7;
    int c   = tid & 127;
    w2b[tid] = (__bf16)((o < NTAP) ? w2[o * CMID + c] : 0.0f);
}

// ---------------------------------------------------------------------------
// Kernel 4: dilated 7x7 conv 256->128, bf16 WMMA implicit GEMM.
// One block per (b,h) row: M = 64 padded pixels, N = 128.
// 8 waves: wave -> N-tile; 4 M-subtiles => 4 accumulators per wave.
// A staged per tap in double-buffered LDS (2 x 32KB), async when available.
// 49 taps x 8 K-chunks x 4 subtiles = 1568 v_wmma per wave.
// ---------------------------------------------------------------------------
__global__ __launch_bounds__(256) void conv1_wmma_bf16(
    const __bf16* __restrict__ xh, const __bf16* __restrict__ w1b,
    const float* __restrict__ bias1, __bf16* __restrict__ k1b)
{
    __shared__ __bf16 smem[2][MBLK * CIN];        // 2 x 32KB

    const int tid     = threadIdx.x;
    const int lane    = tid & 31;
    const int wave    = tid >> 5;
    const int mrow    = lane & 15;                // A row / B,C,D column
    const int hiHalf  = lane >> 4;
    const int h       = blockIdx.x % H;
    const int b       = blockIdx.x / H;

    const int ocol = wave * 16 + mrow;            // output channel (N column)
    const int aoff = hiHalf * 8;                  // 16-bit A layout K sub-offset
    const int boff = hiHalf * 8;                  // 16-bit B layout K sub-offset

    // Padded-row base: window starts at ww = 2 (i.e. w = -6) for tap j=0.
    const __bf16* blockBase =
        xh + ((size_t)(b * HP + h) * WP + 2) * CIN;

    // tap t = 7*i + j  ->  global element offset (2*i*WP + 2*j)*CIN
    auto tapSrc = [&](int t) -> const __bf16* {
        const int i = (t * 37) >> 8;              // t/7 for t < 56
        const int j = t - i * 7;
        return blockBase + (size_t)(2 * i * WP + 2 * j) * CIN;
    };

    v8f acc0 = {}, acc1 = {}, acc2 = {}, acc3 = {};

    const __bf16* brow = w1b + (size_t)ocol * CIN;          // advances per tap
    constexpr size_t BROW_STEP = (size_t)CMID * CIN;        // 32768 elements

    stage_tile(tapSrc(0), &smem[0][0], tid);

    for (int t = 0; t < NTAP; ++t) {
        const int cur = t & 1;
        if (t + 1 < NTAP) {
            stage_tile(tapSrc(t + 1), &smem[1 - cur][0], tid);
            __builtin_prefetch(brow + BROW_STEP, 0, 1);     // next tap weights
            wait_async_lds<8>();                            // tap t complete
        } else {
            wait_async_lds<0>();
        }
        __syncthreads();

        const __bf16* bsrc = brow + boff;
#pragma unroll
        for (int kc = 0; kc < 8; ++kc) {
            const int c0 = kc * 32;

            // B fragment: lane n = column, K 8h..8h+7 and 16+8h..16+8h+7
            v8bf bl = *(const v8bf*)(bsrc + c0);
            v8bf bh = *(const v8bf*)(bsrc + c0 + 16);
            v16bf bfrag;
#pragma unroll
            for (int e = 0; e < 8; ++e) { bfrag[e] = bl[e]; bfrag[8 + e] = bh[e]; }

            // 4 M-subtiles from LDS; A layout per ISA 7.12.2
#pragma unroll
            for (int s = 0; s < 4; ++s) {
                const __bf16* arow = &smem[cur][(s * 16 + mrow) * CIN + c0 + aoff];
                v8bf al = *(const v8bf*)(arow);
                v8bf ah = *(const v8bf*)(arow + 16);
                v16bf afrag;
#pragma unroll
                for (int e = 0; e < 8; ++e) { afrag[e] = al[e]; afrag[8 + e] = ah[e]; }

                v8f* accp = (s == 0) ? &acc0 : (s == 1) ? &acc1 : (s == 2) ? &acc2 : &acc3;
                *accp = __builtin_amdgcn_wmma_f32_16x16x32_bf16(
                    false, afrag, false, bfrag, (short)0, *accp, false, false);
            }
        }
        brow += BROW_STEP;
        __syncthreads();
    }

    // Epilogue: bias + ReLU; store only real pixels (w < 56).
    // C/D layout: VGPR r -> M=r (lanes 0-15) / M=8+r (lanes 16-31), N=lane&15
    const float bv = bias1[ocol];
    const size_t rowPix = (size_t)b * HW + (size_t)h * W;
#pragma unroll
    for (int s = 0; s < 4; ++s) {
        const v8f acc = (s == 0) ? acc0 : (s == 1) ? acc1 : (s == 2) ? acc2 : acc3;
#pragma unroll
        for (int r = 0; r < 8; ++r) {
            const int w = s * 16 + r + hiHalf * 8;
            if (w < W) {
                float v = acc[r] + bv;
                v = v > 0.0f ? v : 0.0f;
                k1b[(rowPix + w) * CMID + ocol] = (__bf16)v;
            }
        }
    }
}

// ---------------------------------------------------------------------------
// Kernel 5: 1x1 conv 128->49 (padded N=64) WMMA + BN(inference) + ReLU.
// ---------------------------------------------------------------------------
__global__ __launch_bounds__(128) void conv2_wmma_bf16(
    const __bf16* __restrict__ k1b, const __bf16* __restrict__ w2b,
    const float* __restrict__ b2, const float* __restrict__ gamma,
    const float* __restrict__ beta, const float* __restrict__ rmean,
    const float* __restrict__ rvar, float* __restrict__ logits)
{
    const int lane    = threadIdx.x & 31;
    const int wave    = threadIdx.x >> 5;
    const int mrow    = lane & 15;
    const int hiHalf  = lane >> 4;
    const int pixBase = blockIdx.x * 16;

    const int p    = pixBase + mrow;
    const int ocol = wave * 16 + mrow;
    const int aoff = hiHalf * 8;
    const int boff = hiHalf * 8;

    const __bf16* arow = k1b + (size_t)p * CMID;
    const __bf16* brow = w2b + (size_t)ocol * CMID;

    v8f acc = {};
#pragma unroll
    for (int kc = 0; kc < 4; ++kc) {
        const int c0 = kc * 32;
        v8bf al = *(const v8bf*)(arow + c0 + aoff);
        v8bf ah = *(const v8bf*)(arow + c0 + 16 + aoff);
        v16bf afrag;
#pragma unroll
        for (int e = 0; e < 8; ++e) { afrag[e] = al[e]; afrag[8 + e] = ah[e]; }
        v8bf bl = *(const v8bf*)(brow + c0 + boff);
        v8bf bh = *(const v8bf*)(brow + c0 + boff + 16);
        v16bf bfrag;
#pragma unroll
        for (int e = 0; e < 8; ++e) { bfrag[e] = bl[e]; bfrag[8 + e] = bh[e]; }
        acc = __builtin_amdgcn_wmma_f32_16x16x32_bf16(
            false, afrag, false, bfrag, (short)0, acc, false, false);
    }

    const int   oc = ocol < NTAP ? ocol : NTAP - 1;
    const float bb = b2[oc];
    const float mn = rmean[oc];
    const float iv = gamma[oc] * rsqrtf(rvar[oc] + BN_EPS);
    const float bt = beta[oc];
#pragma unroll
    for (int r = 0; r < 8; ++r) {
        const int pr = pixBase + r + hiHalf * 8;
        float v = (acc[r] + bb - mn) * iv + bt;
        v = v > 0.0f ? v : 0.0f;
        if (ocol >= NTAP) v = 0.0f;
        logits[(size_t)pr * NATT + ocol] = v;
    }
}

// ---------------------------------------------------------------------------
// Kernel 6: in-place softmax over the 49 attention channels per pixel.
// ---------------------------------------------------------------------------
__global__ __launch_bounds__(256) void softmax49(float* __restrict__ attn)
{
    int p = blockIdx.x * 256 + threadIdx.x;
    if (p >= NPIX) return;
    float* L = attn + (size_t)p * NATT;
    float m = -1e30f;
    for (int t = 0; t < NTAP; ++t) m = fmaxf(m, L[t]);
    float s = 0.0f;
    for (int t = 0; t < NTAP; ++t) s += __expf(L[t] - m);
    const float inv = 1.0f / s;
    for (int t = 0; t < NTAP; ++t) L[t] = __expf(L[t] - m) * inv;
}

// ---------------------------------------------------------------------------
// Kernel 7: out[b,c,h,w] = sum_t attn[p][t] * x[b,c,h+2i-6,w+2j-6] (zero pad).
// One block per (b,c,h) row; threads along w => coalesced tap reads.
// ---------------------------------------------------------------------------
__global__ __launch_bounds__(64) void aggregate49(
    const float* __restrict__ x, const float* __restrict__ attn,
    float* __restrict__ out)
{
    const int w   = threadIdx.x;
    const int blk = blockIdx.x;                 // b*(CIN*H) + c*H + h
    const int h   = blk % H;
    const int c   = (blk / H) % CIN;
    const int b   = blk / (H * CIN);
    if (w >= W) return;

    const size_t p  = ((size_t)b * H + h) * W + w;
    const float* A  = attn + p * NATT;
    const float* xc = x + ((size_t)b * CIN + c) * HW;

    float acc = 0.0f;
#pragma unroll
    for (int t = 0; t < NTAP; ++t) {
        const int ti = t / 7, tj = t % 7;
        const int h2 = h + 2 * ti - 6;
        const int w2 = w + 2 * tj - 6;
        if (h2 >= 0 && h2 < H && w2 >= 0 && w2 < W)
            acc += A[t] * xc[h2 * W + w2];
    }
    out[(((size_t)b * CIN + c) * H + h) * W + w] = acc;
}

// ---------------------------------------------------------------------------
extern "C" void kernel_launch(void* const* d_in, const int* in_sizes, int n_in,
                              void* d_out, int out_size, void* d_ws, size_t ws_size,
                              hipStream_t stream)
{
    const float* x     = (const float*)d_in[0];
    const float* w1    = (const float*)d_in[1];
    const float* b1    = (const float*)d_in[2];
    const float* w2    = (const float*)d_in[3];
    const float* b2    = (const float*)d_in[4];
    const float* gamma = (const float*)d_in[5];
    const float* beta  = (const float*)d_in[6];
    const float* rmean = (const float*)d_in[7];
    const float* rvar  = (const float*)d_in[8];
    float* out = (float*)d_out;

    char* ws = (char*)d_ws;
    __bf16* xh     = (__bf16*)(ws + OFF_XH);
    __bf16* w1b    = (__bf16*)(ws + OFF_W1B);
    __bf16* w2b    = (__bf16*)(ws + OFF_W2B);
    __bf16* k1b    = (__bf16*)(ws + OFF_K1B);
    float*  attn   = (float*)(ws + OFF_LOG);    // logits, softmaxed in place

    // Stage 0: precision/layout packing
    pack_x_halo_bf16<<<(BATCH * HP * WP * CIN) / 256, 256, 0, stream>>>(x, xh);
    pack_w1_bf16<<<(NTAP * CMID * CIN) / 256, 256, 0, stream>>>(w1, w1b);
    pack_w2_bf16<<<(NATT * CMID) / 256, 256, 0, stream>>>(w2, w2b);

    // Stage 1: dilated 7x7 conv 256->128 (WMMA bf16), bias+ReLU fused
    conv1_wmma_bf16<<<BATCH * H, 256, 0, stream>>>(xh, w1b, b1, k1b);

    // Stage 2: 1x1 conv 128->49 (WMMA bf16) + BN + ReLU
    conv2_wmma_bf16<<<NPIX / 16, 128, 0, stream>>>(k1b, w2b, b2, gamma, beta,
                                                   rmean, rvar, attn);

    // Stage 3: softmax over 49 channels (in place)
    softmax49<<<(NPIX + 255) / 256, 256, 0, stream>>>(attn);

    // Stage 4: attention-weighted 49-tap aggregation on original f32 x
    aggregate49<<<BATCH * CIN * H, 64, 0, stream>>>(x, attn, out);
}
